// VGCNEmbeddings_77034533421632
// MI455X (gfx1250) — compile-verified, matrix operand order checked by reading, not run on hardware
//
#include <hip/hip_runtime.h>
#include <hip/hip_bf16.h>

// ---------------- problem constants (match reference) ----------------
#define VOCAB 30522
#define DM    768          // model dim D
#define PMAX  512
#define BB    32           // batch
#define SS    512          // seq len
#define GG    512          // graph vocab per sample
#define GDIM  16           // per-graph feature dim
#define KIN   816          // D + 3*GD
#define KP    832          // K padded to 26*32 for WMMA K-loop
#define MTOT  (BB*SS)      // 16384 rows
#define LN_EPS 1e-12f

typedef __attribute__((ext_vector_type(16))) _Float16 v16h;
typedef __attribute__((ext_vector_type(8)))  _Float16 v8h;
typedef __attribute__((ext_vector_type(8)))  float    v8f;

// ---------------- K1: adjacency column sums (3 graphs) ----------------
// colsum[gr][b][h] = sum_g adj[b][g][h]   (pure streaming, 100 MB total)
__global__ void __launch_bounds__(256) colsum_kernel(
    const float* __restrict__ a0, const float* __restrict__ a1,
    const float* __restrict__ a2, float* __restrict__ colsum)
{
    const float* adj = (blockIdx.y == 0) ? a0 : (blockIdx.y == 1) ? a1 : a2;
    int idx = blockIdx.x * 256 + threadIdx.x;      // b*G + h
    int b = idx >> 9;
    int h = idx & (GG - 1);
    const float* p = adj + (size_t)b * GG * GG + h;
    float s = 0.f;
    #pragma unroll 4
    for (int g = 0; g < GG; ++g) s += p[(size_t)g * GG];
    colsum[blockIdx.y * (BB * GG) + idx] = s;
}

// ---------------- K2: gvec[gr][b][d] = (1/G) sum_h colsum*emb_row ------
__global__ void __launch_bounds__(256) gvec_kernel(
    const float* __restrict__ colsum,
    const int* __restrict__ gv0, const int* __restrict__ gv1,
    const int* __restrict__ gv2,
    const float* __restrict__ wemb, float* __restrict__ gvec)
{
    int gr = blockIdx.z;
    const int* gv = (gr == 0) ? gv0 : (gr == 1) ? gv1 : gv2;
    int b = blockIdx.y;
    int d = blockIdx.x * 256 + threadIdx.x;
    const float* cs   = colsum + gr * BB * GG + b * GG;
    const int*   rows = gv + b * GG;
    float acc = 0.f;
    for (int h = 0; h < GG; ++h) {
        acc += cs[h] * wemb[(size_t)rows[h] * DM + d];
    }
    gvec[((size_t)gr * BB + b) * DM + d] = acc * (1.f / (float)GG);
}

// ---------------- K3: graph features = relu(gvec @ W.T + b) -----------
__global__ void __launch_bounds__(256) gfeat_kernel(
    const float* __restrict__ gvec,
    const float* __restrict__ W0, const float* __restrict__ bi0,
    const float* __restrict__ W1, const float* __restrict__ bi1,
    const float* __restrict__ W2, const float* __restrict__ bi2,
    float* __restrict__ gfeat)
{
    int idx = blockIdx.x * 256 + threadIdx.x;      // b*48 + c
    if (idx >= BB * 48) return;
    int b = idx / 48, c = idx % 48;
    int gr = c >> 4, j = c & 15;
    const float* W  = (gr == 0) ? W0  : (gr == 1) ? W1  : W2;
    const float* bi = (gr == 0) ? bi0 : (gr == 1) ? bi1 : bi2;
    const float* g  = gvec + ((size_t)gr * BB + b) * DM;
    float acc = bi[j];
    #pragma unroll 4
    for (int d = 0; d < DM; ++d) acc += g[d] * W[j * DM + d];
    gfeat[idx] = fmaxf(acc, 0.f);
}

// ---------------- K4: pack W_proj -> f16 [768][KP] (zero pad K) -------
__global__ void __launch_bounds__(256) packw_kernel(
    const float* __restrict__ wproj, _Float16* __restrict__ wh)
{
    int idx = blockIdx.x * 256 + threadIdx.x;      // n*KP + k
    int n = idx / KP, k = idx - n * KP;
    float v = (k < KIN) ? wproj[n * KIN + k] : 0.f;
    wh[idx] = (_Float16)v;
}

// ---------------- K5: build x -> f16 [MTOT][KP] -----------------------
// k<768: word_emb[ids[m]][k] + pos_emb[s][k]; 768<=k<816: gfeat; else 0
__global__ void __launch_bounds__(256) buildx_kernel(
    const int* __restrict__ ids, const float* __restrict__ wemb,
    const float* __restrict__ pemb, const float* __restrict__ gfeat,
    _Float16* __restrict__ xh)
{
    unsigned idx = blockIdx.x * 256u + threadIdx.x;  // m*KP + k (max ~13.6M)
    int m = idx / KP;
    int k = idx - (unsigned)m * KP;
    int b = m >> 9;
    int s = m & (SS - 1);
    float v;
    if (k < DM)       v = wemb[(size_t)ids[m] * DM + k] + pemb[s * DM + k];
    else if (k < KIN) v = gfeat[b * 48 + (k - DM)];
    else              v = 0.f;
    xh[idx] = (_Float16)v;
}

// ---------------- K6: fused WMMA GEMM + bias + LayerNorm --------------
// Block = 256 threads (8 waves), computes 16 output rows x full N=768.
// Wave w owns a 16x96 strip (6 WMMA accumulator tiles).
// Operand VGPR layout per ISA 7.12.2 (16-bit A 16x32 / B 32x16):
//   lanes 0-15  hold K {k0..k0+7, k0+16..k0+23}
//   lanes 16-31 hold K {k0+8..k0+15, k0+24..k0+31}
// Both x (row m, K-contig) and W (row n, K-contig) are K-major, so each
// lane issues two global_load_b128 per operand per K-step.
__global__ void __launch_bounds__(256) gemm_ln_kernel(
    const _Float16* __restrict__ xh,    // [MTOT][KP]
    const _Float16* __restrict__ wh,    // [768][KP]
    const float* __restrict__ bproj,    // [768]
    const float* __restrict__ gamma,
    const float* __restrict__ beta,
    float* __restrict__ out)            // [MTOT][768]
{
    __shared__ float ytile[16][776];    // 16 rows x 768 (+pad) = 49.7 KB LDS

    const int tid    = threadIdx.x;
    const int wave   = tid >> 5;
    const int lane   = tid & 31;
    const int lane16 = lane & 15;
    const int khalf  = (lane >> 4) << 3;   // 0 or 8: K sub-block per half-wave

    const int mBase = blockIdx.x * 16;
    const int nBase = wave * 96;

    const _Float16* pA  = xh + (size_t)(mBase + lane16) * KP + khalf;
    const _Float16* pB0 = wh + (size_t)(nBase + lane16) * KP + khalf;

    v8f acc[6] = {};

    for (int kk = 0; kk < KP; kk += 32) {
        if (kk + 32 < KP)
            __builtin_prefetch(pA + kk + 32, 0, 1);   // global_prefetch_b8

        v8h alo = *(const v8h*)(pA + kk);
        v8h ahi = *(const v8h*)(pA + kk + 16);
        v16h a = __builtin_shufflevector(alo, ahi,
                 0,1,2,3,4,5,6,7,8,9,10,11,12,13,14,15);

        #pragma unroll
        for (int t = 0; t < 6; ++t) {
            const _Float16* pB = pB0 + (size_t)t * 16 * KP + kk;
            v8h blo = *(const v8h*)(pB);
            v8h bhi = *(const v8h*)(pB + 16);
            v16h bmat = __builtin_shufflevector(blo, bhi,
                        0,1,2,3,4,5,6,7,8,9,10,11,12,13,14,15);
            acc[t] = __builtin_amdgcn_wmma_f32_16x16x32_f16(
                false, a, false, bmat, (short)0, acc[t], false, false);
        }
    }

    // C/D layout: VGPR r -> M = r + (lane<16 ? 0 : 8), N = lane&15 (per tile)
    #pragma unroll
    for (int t = 0; t < 6; ++t) {
        int n = nBase + t * 16 + lane16;
        float bias = bproj[n];
        #pragma unroll
        for (int r = 0; r < 8; ++r) {
            ytile[r + khalf][n] = acc[t][r] + bias;
        }
    }
    __syncthreads();

    // LayerNorm: 16 groups of 16 threads, one row per group
    const int row = tid >> 4;
    const int gl  = tid & 15;
    float s = 0.f, sq = 0.f;
    #pragma unroll 8
    for (int j = gl; j < DM; j += 16) {
        float v = ytile[row][j];
        s += v; sq += v * v;
    }
    #pragma unroll
    for (int off = 8; off >= 1; off >>= 1) {
        s  += __shfl_xor(s,  off, 16);
        sq += __shfl_xor(sq, off, 16);
    }
    float mu   = s  * (1.f / (float)DM);
    float var  = sq * (1.f / (float)DM) - mu * mu;
    float rstd = rsqrtf(var + LN_EPS);

    float* orow = out + (size_t)(mBase + row) * DM;
    #pragma unroll 8
    for (int j = gl; j < DM; j += 16) {
        orow[j] = (ytile[row][j] - mu) * rstd * gamma[j] + beta[j];
    }
}

// ---------------- host-side launcher ----------------------------------
extern "C" void kernel_launch(void* const* d_in, const int* in_sizes, int n_in,
                              void* d_out, int out_size, void* d_ws, size_t ws_size,
                              hipStream_t stream)
{
    const int*   input_ids = (const int*)  d_in[0];
    const float* adj_cooc  = (const float*)d_in[1];
    const int*   gv_cooc   = (const int*)  d_in[2];
    const float* adj_het   = (const float*)d_in[3];
    const int*   gv_het    = (const int*)  d_in[4];
    const float* adj_isg   = (const float*)d_in[5];
    const int*   gv_isg    = (const int*)  d_in[6];
    const float* word_emb  = (const float*)d_in[7];
    const float* pos_emb   = (const float*)d_in[8];
    const float* W_cooc    = (const float*)d_in[9];
    const float* b_cooc    = (const float*)d_in[10];
    const float* W_het     = (const float*)d_in[11];
    const float* b_het     = (const float*)d_in[12];
    const float* W_isg     = (const float*)d_in[13];
    const float* b_isg     = (const float*)d_in[14];
    const float* W_proj    = (const float*)d_in[15];
    const float* b_proj    = (const float*)d_in[16];
    const float* ln_gamma  = (const float*)d_in[17];
    const float* ln_beta   = (const float*)d_in[18];
    float* out = (float*)d_out;

    // workspace layout (bytes); total ~29 MB, fully rewritten every call
    char* ws = (char*)d_ws;
    float*    colsum = (float*)   (ws + 0);                    // 3*B*G f32   = 192 KB
    float*    gvec   = (float*)   (ws + 196608);               // 3*B*D f32   = 288 KB
    float*    gfeat  = (float*)   (ws + 491520);               // B*48 f32    =   6 KB
    _Float16* wh     = (_Float16*)(ws + 497664);               // 768*KP f16  = 1.22 MB
    _Float16* xh     = (_Float16*)(ws + 1775616);              // MTOT*KP f16 =  26 MB

    // K1: adjacency column sums
    colsum_kernel<<<dim3((BB * GG) / 256, 3), 256, 0, stream>>>(
        adj_cooc, adj_het, adj_isg, colsum);

    // K2: weighted embedding sums (mean-pooled GCN output, algebraically folded)
    gvec_kernel<<<dim3(DM / 256, BB, 3), 256, 0, stream>>>(
        colsum, gv_cooc, gv_het, gv_isg, word_emb, gvec);

    // K3: relu(fc) graph features
    gfeat_kernel<<<(BB * 48 + 255) / 256, 256, 0, stream>>>(
        gvec, W_cooc, b_cooc, W_het, b_het, W_isg, b_isg, gfeat);

    // K4: pack W_proj to f16, K padded to 832
    packw_kernel<<<(DM * KP) / 256, 256, 0, stream>>>(W_proj, wh);

    // K5: build fused input rows (emb gather + pos + graph concat) in f16
    buildx_kernel<<<(MTOT * KP) / 256, 256, 0, stream>>>(
        input_ids, word_emb, pos_emb, gfeat, xh);

    // K6: WMMA GEMM + bias + LayerNorm, fused epilogue in LDS
    gemm_ln_kernel<<<MTOT / 16, 256, 0, stream>>>(
        xh, wh, b_proj, ln_gamma, ln_beta, out);
}
